// MipNerf_91328184582839
// MI455X (gfx1250) — matrix operand order repaired and checked by reading, not compiled, not run on hardware
//
#include <hip/hip_runtime.h>
#include <hip/hip_bf16.h>
#include <math.h>

// ---------------------------------------------------------------------------
// MipNeRF two-level pipeline for gfx1250 (MI455X).
// GEMMs use v_wmma_f32_16x16x32_f16 (wave32 WMMA), f16 activations, f32 accum.
// A-fragments register-resident across all N tiles; B-fragments batch-loaded
// per tile so the KS WMMAs issue as one dense accumulate chain.
// ---------------------------------------------------------------------------

typedef _Float16 h16;
typedef h16   v16h __attribute__((ext_vector_type(16)));
typedef float v8f  __attribute__((ext_vector_type(8)));

union Frag { v16h h; uint4 q[2]; };

#define R_  4096
#define NS_ 128
#define P_  (R_ * NS_)        // 524288 points per level
#define PC_ 32768             // points per chunk (P_/PC_ = 16 chunks)

// ---------------------------------------------------------------------------
// Weight conversion: W f32 [K, N] row-major  ->  Wt f16 [N, Kpad] (transposed,
// zero padded in K). B-fragment loads then read contiguous K pairs per lane.
// ---------------------------------------------------------------------------
__global__ void k_convert_w(const float* __restrict__ W, h16* __restrict__ Wt,
                            int K, int Kpad, int N) {
  int tid = blockIdx.x * blockDim.x + threadIdx.x;
  if (tid >= N * Kpad) return;
  int n = tid / Kpad;
  int k = tid - n * Kpad;
  Wt[tid] = (k < K) ? (h16)W[(long)k * N + n] : (h16)0.0f;
}

// ---------------------------------------------------------------------------
// t_samples init: near + (far-near)*linspace(0,1,NS+1)
// ---------------------------------------------------------------------------
__global__ void k_init_t(const float* __restrict__ nearp,
                         const float* __restrict__ farp,
                         float* __restrict__ T) {
  int tid = blockIdx.x * blockDim.x + threadIdx.x;
  if (tid >= R_ * (NS_ + 1)) return;
  int r = tid / (NS_ + 1);
  int i = tid - r * (NS_ + 1);
  T[tid] = nearp[r] + (farp[r] - nearp[r]) * ((float)i * (1.0f / (float)NS_));
}

// ---------------------------------------------------------------------------
// View encoding: [v(3), sin(v*2^d)(12), cos(v*2^d)(12)] -> 27 floats per ray
// ---------------------------------------------------------------------------
__global__ void k_view_enc(const float* __restrict__ vd, float* __restrict__ VE) {
  int r = blockIdx.x * blockDim.x + threadIdx.x;
  if (r >= R_) return;
  float v[3] = {vd[r * 3], vd[r * 3 + 1], vd[r * 3 + 2]};
  float* e = VE + (long)r * 27;
  e[0] = v[0]; e[1] = v[1]; e[2] = v[2];
  for (int d = 0; d < 4; ++d) {
    float sc = (float)(1 << d);
    for (int a = 0; a < 3; ++a) {
      float xb = v[a] * sc;
      e[3 + d * 3 + a]  = sinf(xb);
      e[15 + d * 3 + a] = cosf(xb);
    }
  }
}

// ---------------------------------------------------------------------------
// Conical-gaussian cast + integrated positional encoding -> enc f16 [PC_, 96]
// ---------------------------------------------------------------------------
__global__ void k_encode(const float* __restrict__ T,
                         const float* __restrict__ org,
                         const float* __restrict__ dir,
                         const float* __restrict__ radii,
                         h16* __restrict__ enc, int chunkStart) {
  int p  = blockIdx.x * blockDim.x + threadIdx.x;
  int gp = chunkStart + p;
  int r  = gp >> 7;          // / NS_
  int s  = gp & (NS_ - 1);
  float t0 = T[r * (NS_ + 1) + s];
  float t1 = T[r * (NS_ + 1) + s + 1];
  float mu = 0.5f * (t0 + t1), hw = 0.5f * (t1 - t0);
  float mu2 = mu * mu, hw2 = hw * hw;
  float denom = 3.0f * mu2 + hw2;
  float t_mean = mu + 2.0f * mu * hw2 / denom;
  float t_var  = hw2 * (1.0f / 3.0f)
               - (4.0f / 15.0f) * (hw2 * hw2 * (12.0f * mu2 - hw2)) / (denom * denom);
  float rad = radii[r];
  float r_var = rad * rad * (mu2 * 0.25f + (5.0f / 12.0f) * hw2
                             - (4.0f / 15.0f) * hw2 * hw2 / denom);
  float d0 = dir[r * 3], d1 = dir[r * 3 + 1], d2 = dir[r * 3 + 2];
  float dms = fmaxf(1e-10f, d0 * d0 + d1 * d1 + d2 * d2);
  float dv[3] = {d0, d1, d2};
  float mean[3], cov[3];
  for (int a = 0; a < 3; ++a) {
    mean[a] = dv[a] * t_mean + org[r * 3 + a];
    float dd = dv[a] * dv[a];
    cov[a] = t_var * dd + r_var * (1.0f - dd / dms);
  }
  h16* e = enc + (long)p * 96;
  float sc = 1.0f;
  for (int d = 0; d < 16; ++d) {
    for (int a = 0; a < 3; ++a) {
      float y  = mean[a] * sc;
      float yv = cov[a] * sc * sc;
      float g  = expf(-0.5f * yv);
      e[d * 3 + a]      = (h16)(g * sinf(y));
      e[48 + d * 3 + a] = (h16)(g * cosf(y));
    }
    sc *= 2.0f;
  }
}

// ---------------------------------------------------------------------------
// WMMA GEMM:  Y[rows, NCOLS] = act(X[rows, KPAD(ldx)] @ Wt^T + bias)
//   X  : f16, leading dim ldx;  Wt : f16 [NCOLS, KPAD] (pre-transposed).
//   block = 256 threads = 8 wave32; each wave owns 16 rows.
//   All A fragments for the wave's 16xKPAD strip are loaded ONCE into
//   registers; per N tile the KS B fragments are batch-loaded (one clause,
//   one wait), then the KS WMMAs issue back-to-back.
// A-fragment layout (16-bit A 16x32, ISA 7.12.2): lane m=L%16;
//   lanes 0-15  : V0..3 -> K 0..7,  V4..7 -> K16..23
//   lanes 16-31 : V0..3 -> K 8..15, V4..7 -> K24..31
// B-fragment (16-bit B 32x16): lane n=L%16; same K order as A (lanes0-15 K0..7,K16..23; lanes16-31 K8..15,K24..31)
// ---------------------------------------------------------------------------
template <int KPAD, int NCOLS, int RELU>
__global__ __launch_bounds__(256) void k_gemm_t(
    const h16* __restrict__ X, int ldx,
    const h16* __restrict__ Wt,
    const float* __restrict__ bias,
    h16* __restrict__ Y, int ldy) {
  constexpr int KS = KPAD / 32;
  constexpr int NT = NCOLS / 16;
  int lane = threadIdx.x & 31;
  int wv   = threadIdx.x >> 5;
  int lm   = lane & 15;
  int hi   = lane >> 4;
  long row0 = (long)blockIdx.x * 128 + wv * 16;
  const h16* xrow = X + (row0 + lm) * (long)ldx;

  Frag a[KS];
#pragma unroll
  for (int kk = 0; kk < KS; ++kk) {
    a[kk].q[0] = *(const uint4*)(xrow + kk * 32 + hi * 8);
    a[kk].q[1] = *(const uint4*)(xrow + kk * 32 + 16 + hi * 8);
  }

  for (int nt = 0; nt < NT; ++nt) {
    const h16* wrow = Wt + (long)(nt * 16 + lm) * KPAD;
    if (nt + 1 < NT)
      __builtin_prefetch(wrow + (long)16 * KPAD, 0, 3);  // global_prefetch_b8 (near)
    // Batch-load all B fragments for this tile: one load clause, one wait.
    Frag b[KS];
#pragma unroll
    for (int kk = 0; kk < KS; ++kk) {
      b[kk].q[0] = *(const uint4*)(wrow + kk * 32 + hi * 8);
      b[kk].q[1] = *(const uint4*)(wrow + kk * 32 + 16 + hi * 8);
    }
    v8f acc = {0.f, 0.f, 0.f, 0.f, 0.f, 0.f, 0.f, 0.f};
#pragma unroll
    for (int kk = 0; kk < KS; ++kk) {
      acc = __builtin_amdgcn_wmma_f32_16x16x32_f16(
          /*neg_a=*/false, a[kk].h, /*neg_b=*/false, b[kk].h,
          /*c_mod=*/(short)0, acc, /*reuse_a=*/false, /*reuse_b=*/false);
    }
    float bv = bias[nt * 16 + lm];
#pragma unroll
    for (int v = 0; v < 8; ++v) {
      float y = acc[v] + bv;
      if (RELU) y = fmaxf(y, 0.0f);
      Y[(row0 + v + hi * 8) * (long)ldy + nt * 16 + lm] = (h16)y;
    }
  }
}

// Skip connection: copy enc (96) into columns [256,352) of actA
__global__ void k_concat_enc(const h16* __restrict__ enc, h16* __restrict__ actA) {
  int tid = blockIdx.x * blockDim.x + threadIdx.x;
  int p = tid / 96;
  int j = tid - p * 96;
  actA[(long)p * 352 + 256 + j] = enc[(long)p * 96 + j];
}

// Concat view encoding (27) + zero pad (5) into cond input cols [256,288)
__global__ void k_cond_cat(const float* __restrict__ VE, h16* __restrict__ cin,
                           int chunkStart) {
  int p = blockIdx.x * blockDim.x + threadIdx.x;
  int r = (chunkStart + p) >> 7;
  h16* c = cin + (long)p * 288 + 256;
  for (int j = 0; j < 27; ++j) c[j] = (h16)VE[(long)r * 27 + j];
  for (int j = 27; j < 32; ++j) c[j] = (h16)0.0f;
}

// density = softplus(x256 . w + b - 1)
__global__ void k_density(const h16* __restrict__ X, const float* __restrict__ w,
                          const float* __restrict__ b, float* __restrict__ dens,
                          int chunkStart) {
  int p = blockIdx.x * blockDim.x + threadIdx.x;
  const h16* x = X + (long)p * 352;
  float acc = 0.f;
  for (int i = 0; i < 256; ++i) acc += (float)x[i] * w[i];
  float raw = acc + b[0] - 1.0f;
  dens[chunkStart + p] = (raw > 20.f) ? raw : log1pf(expf(raw));
}

// rgb = sigmoid(x128 @ W[128,3] + b)*(1+2*pad) - pad
__global__ void k_rgb(const h16* __restrict__ X, const float* __restrict__ w,
                      const float* __restrict__ b, float* __restrict__ rgb,
                      int chunkStart) {
  int p = blockIdx.x * blockDim.x + threadIdx.x;
  const h16* x = X + (long)p * 352;
  for (int c = 0; c < 3; ++c) {
    float acc = b[c];
    for (int i = 0; i < 128; ++i) acc += (float)x[i] * w[i * 3 + c];
    float sg = 1.0f / (1.0f + expf(-acc));
    rgb[(long)(chunkStart + p) * 3 + c] = sg * 1.002f - 0.001f;
  }
}

// Volumetric rendering (one thread / ray, sequential transmittance scan)
__global__ void k_render(const float* __restrict__ T, const float* __restrict__ dir,
                         const float* __restrict__ dens, const float* __restrict__ rgb,
                         float* __restrict__ wbuf, float* __restrict__ out, int level) {
  int r = blockIdx.x * blockDim.x + threadIdx.x;
  if (r >= R_) return;
  float d0 = dir[r * 3], d1 = dir[r * 3 + 1], d2 = dir[r * 3 + 2];
  float dn = sqrtf(d0 * d0 + d1 * d1 + d2 * d2);
  float accum = 0.f, cr = 0.f, cg = 0.f, cb = 0.f, wsum = 0.f;
  for (int s = 0; s < NS_; ++s) {
    float t0 = T[r * (NS_ + 1) + s];
    float t1 = T[r * (NS_ + 1) + s + 1];
    float dd = dens[r * NS_ + s] * (t1 - t0) * dn;
    float trans = expf(-accum);
    float alpha = 1.0f - expf(-dd);
    float w = alpha * trans;
    accum += dd;
    long q = (long)(r * NS_ + s) * 3;
    cr += w * rgb[q]; cg += w * rgb[q + 1]; cb += w * rgb[q + 2];
    wsum += w;
    wbuf[r * NS_ + s] = w;
  }
  float bg = 1.0f - wsum;
  out[(long)level * R_ * 3 + r * 3 + 0] = cr + bg;
  out[(long)level * R_ * 3 + r * 3 + 1] = cg + bg;
  out[(long)level * R_ * 3 + r * 3 + 2] = cb + bg;
}

// Inverse-CDF resampling (one thread / ray; monotone u -> two-pointer search)
__global__ void k_resample(const float* __restrict__ Tin, const float* __restrict__ wbuf,
                           float* __restrict__ Tout) {
  int r = blockIdx.x * blockDim.x + threadIdx.x;
  if (r >= R_) return;
  float w[NS_];
  float sum = 0.f;
  for (int i = 0; i < NS_; ++i) {
    int im1 = (i - 1) < 0 ? 0 : i - 1;
    int ip1 = (i + 1) > (NS_ - 1) ? (NS_ - 1) : i + 1;
    float a = fmaxf(wbuf[r * NS_ + im1], wbuf[r * NS_ + i]);
    float b = fmaxf(wbuf[r * NS_ + i], wbuf[r * NS_ + ip1]);
    w[i] = 0.5f * (a + b) + 0.01f;
    sum += w[i];
  }
  float pad = fmaxf(0.f, 1e-5f - sum);
  sum += pad;
  pad *= (1.0f / (float)NS_);
  float inv = 1.0f / sum;
  float cdf[NS_ + 1];
  cdf[0] = 0.f;
  float c = 0.f;
  for (int i = 1; i < NS_; ++i) {
    c += (w[i - 1] + pad) * inv;
    cdf[i] = fminf(1.0f, c);
  }
  cdf[NS_] = 1.0f;
  const float step = (1.0f - 1.1920929e-7f) / (float)NS_;
  int idx = 0;
  for (int j = 0; j <= NS_; ++j) {
    float u = (float)j * step;
    while (idx + 1 < NS_ + 1 && cdf[idx + 1] <= u) ++idx;
    float c0 = cdf[idx], c1 = cdf[idx + 1];
    float b0 = Tin[r * (NS_ + 1) + idx], b1 = Tin[r * (NS_ + 1) + idx + 1];
    float t = (u - c0) / fmaxf(c1 - c0, 1e-10f);
    t = fminf(fmaxf(t, 0.0f), 1.0f);
    Tout[r * (NS_ + 1) + j] = b0 + t * (b1 - b0);
  }
}

// ---------------------------------------------------------------------------
extern "C" void kernel_launch(void* const* d_in, const int* in_sizes, int n_in,
                              void* d_out, int out_size, void* d_ws, size_t ws_size,
                              hipStream_t stream) {
  (void)n_in; (void)out_size; (void)ws_size;
  const float* origins    = (const float*)d_in[0];
  const float* directions = (const float*)d_in[1];
  const float* viewdirs   = (const float*)d_in[2];
  const float* radii      = (const float*)d_in[3];
  const float* nearp      = (const float*)d_in[4];
  const float* farp       = (const float*)d_in[5];

  // Param ordering: JAX pytree alphabetizes dict keys; detect via sizes.
  const float *tw[8], *tb[8], *dw, *db, *bw, *bb, *cw, *cb, *rw, *rb;
  if (in_sizes[6] == 96 * 256) {           // insertion order: trunk first
    for (int i = 0; i < 8; ++i) {
      tw[i] = (const float*)d_in[6 + 2 * i];
      tb[i] = (const float*)d_in[7 + 2 * i];
    }
    dw = (const float*)d_in[22]; db = (const float*)d_in[23];
    bw = (const float*)d_in[24]; bb = (const float*)d_in[25];
    cw = (const float*)d_in[26]; cb = (const float*)d_in[27];
    rw = (const float*)d_in[28]; rb = (const float*)d_in[29];
  } else {                                  // alphabetical: bott, cond, dens, rgb, trunk
    bw = (const float*)d_in[6];  bb = (const float*)d_in[7];
    cw = (const float*)d_in[8];  cb = (const float*)d_in[9];
    dw = (const float*)d_in[10]; db = (const float*)d_in[11];
    rw = (const float*)d_in[12]; rb = (const float*)d_in[13];
    for (int i = 0; i < 8; ++i) {
      tw[i] = (const float*)d_in[14 + 2 * i];
      tb[i] = (const float*)d_in[15 + 2 * i];
    }
  }

  // Workspace carve-out (256B aligned)
  size_t off = 0;
  auto alloc = [&](size_t bytes) -> char* {
    char* p = (char*)d_ws + off;
    off += (bytes + 255) & ~(size_t)255;
    return p;
  };
  const int KT[8] = {96, 256, 256, 256, 256, 352, 256, 256};
  h16* WT[8];
  for (int i = 0; i < 8; ++i) WT[i] = (h16*)alloc((size_t)KT[i] * 256 * 2);
  h16*   WB     = (h16*)alloc((size_t)256 * 256 * 2);     // bottleneck [256,256]
  h16*   WC     = (h16*)alloc((size_t)128 * 288 * 2);     // cond [128,288] padded
  h16*   enc    = (h16*)alloc((size_t)PC_ * 96 * 2);
  h16*   actA   = (h16*)alloc((size_t)PC_ * 352 * 2);
  h16*   actB   = (h16*)alloc((size_t)PC_ * 352 * 2);
  h16*   condin = (h16*)alloc((size_t)PC_ * 288 * 2);
  float* VE     = (float*)alloc((size_t)R_ * 27 * 4);
  float* T0     = (float*)alloc((size_t)R_ * (NS_ + 1) * 4);
  float* T1     = (float*)alloc((size_t)R_ * (NS_ + 1) * 4);
  float* dens   = (float*)alloc((size_t)P_ * 4);
  float* rgbb   = (float*)alloc((size_t)P_ * 12);
  float* wbuf   = (float*)alloc((size_t)P_ * 4);
  float* out    = (float*)d_out;

  // Weight conversion (f32 -> transposed f16, zero-padded K)
  for (int i = 0; i < 8; ++i)
    k_convert_w<<<(KT[i] * 256 + 255) / 256, 256, 0, stream>>>(tw[i], WT[i], KT[i], KT[i], 256);
  k_convert_w<<<(256 * 256 + 255) / 256, 256, 0, stream>>>(bw, WB, 256, 256, 256);
  k_convert_w<<<(128 * 288 + 255) / 256, 256, 0, stream>>>(cw, WC, 283, 288, 128);

  k_init_t<<<(R_ * (NS_ + 1) + 255) / 256, 256, 0, stream>>>(nearp, farp, T0);
  k_view_enc<<<R_ / 256, 256, 0, stream>>>(viewdirs, VE);

  for (int level = 0; level < 2; ++level) {
    const float* T = (level == 0) ? T0 : T1;
    for (int c0 = 0; c0 < P_; c0 += PC_) {
      k_encode<<<PC_ / 256, 256, 0, stream>>>(T, origins, directions, radii, enc, c0);
      dim3 gg(PC_ / 128);
      // Trunk (8 layers, skip-concat after layer index 4)
      k_gemm_t<96, 256, 1><<<gg, 256, 0, stream>>>(enc, 96,  WT[0], tb[0], actA, 352);
      k_gemm_t<256, 256, 1><<<gg, 256, 0, stream>>>(actA, 352, WT[1], tb[1], actB, 352);
      k_gemm_t<256, 256, 1><<<gg, 256, 0, stream>>>(actB, 352, WT[2], tb[2], actA, 352);
      k_gemm_t<256, 256, 1><<<gg, 256, 0, stream>>>(actA, 352, WT[3], tb[3], actB, 352);
      k_gemm_t<256, 256, 1><<<gg, 256, 0, stream>>>(actB, 352, WT[4], tb[4], actA, 352);
      k_concat_enc<<<PC_ * 96 / 256, 256, 0, stream>>>(enc, actA);
      k_gemm_t<352, 256, 1><<<gg, 256, 0, stream>>>(actA, 352, WT[5], tb[5], actB, 352);
      k_gemm_t<256, 256, 1><<<gg, 256, 0, stream>>>(actB, 352, WT[6], tb[6], actA, 352);
      k_gemm_t<256, 256, 1><<<gg, 256, 0, stream>>>(actA, 352, WT[7], tb[7], actB, 352);
      // Heads
      k_density<<<PC_ / 256, 256, 0, stream>>>(actB, dw, db, dens, c0);
      k_gemm_t<256, 256, 0><<<gg, 256, 0, stream>>>(actB, 352, WB, bb, condin, 288);  // bottleneck
      k_cond_cat<<<PC_ / 256, 256, 0, stream>>>(VE, condin, c0);
      k_gemm_t<288, 128, 1><<<gg, 256, 0, stream>>>(condin, 288, WC, cb, actA, 352);  // cond
      k_rgb<<<PC_ / 256, 256, 0, stream>>>(actA, rw, rb, rgbb, c0);
    }
    k_render<<<R_ / 256, 256, 0, stream>>>(T, directions, dens, rgbb, wbuf, out, level);
    if (level == 0)
      k_resample<<<R_ / 256, 256, 0, stream>>>(T0, wbuf, T1);
  }
}
